// AttentionLayer_60790967107900
// MI455X (gfx1250) — matrix operand order repaired and checked by reading, not compiled, not run on hardware
//
#include <hip/hip_runtime.h>

#define BB   4
#define SS   4096
#define DIN  512
#define UU   128

typedef _Float16 half8  __attribute__((ext_vector_type(8)));
typedef _Float16 half16 __attribute__((ext_vector_type(16)));
typedef float    float8 __attribute__((ext_vector_type(8)));

__device__ __forceinline__ float8 wmma_acc(half16 a, half16 b, float8 c) {
    // D = A(16x32 f16) x B(32x16 f16) + C(16x16 f32)
    return __builtin_amdgcn_wmma_f32_16x16x32_f16(
        /*neg_a=*/false, a, /*neg_b=*/false, b,
        /*c_mod=*/(short)0, c, /*reuse_a=*/false, /*reuse_b=*/false);
}

// A-fragment (16x32 f16) from row-major buffer, leading dim `ld` (halves).
__device__ __forceinline__ half16 load_a_frag(const _Float16* base, int ld, int lo, int hi) {
    const half8 a0 = *(const half8*)(base + lo * ld + hi * 16);
    const half8 a1 = *(const half8*)(base + lo * ld + hi * 16 + 8);
    half16 f;
#pragma unroll
    for (int i = 0; i < 8; ++i) { f[i] = a0[i]; f[i + 8] = a1[i]; }
    return f;
}

// Async 16B global->LDS transfer per lane (CDNA5, ASYNCcnt-tracked).
__device__ __forceinline__ void async_b128(const _Float16* g, unsigned lds_off) {
    asm volatile("global_load_async_to_lds_b128 %0, %1, off"
                 :: "v"(lds_off), "v"(g) : "memory");
}
__device__ __forceinline__ void wait_async0() {
    asm volatile("s_wait_asynccnt 0x0" ::: "memory");
}
__device__ __forceinline__ unsigned lds_off(const void* p) {
    return (unsigned)(uintptr_t)p;   // low 32 bits of generic shared ptr = LDS offset
}

// ---------------------------------------------------------------------------
// Kernel 0: one-shot weight transpose+convert: W[512][128] f32 -> Wt[3][128][512] f16.
// ---------------------------------------------------------------------------
__global__ __launch_bounds__(256) void w_transpose_kernel(
    const float* __restrict__ Wq, const float* __restrict__ Wk,
    const float* __restrict__ Wv, _Float16* __restrict__ Wt)
{
    int idx = blockIdx.x * 256 + threadIdx.x;       // 3 * 65536 total
    int mat = idx >> 16;
    int e   = idx & 65535;
    int n   = e >> 9;                                // 0..127
    int k   = e & 511;                               // 0..511
    const float* W = (mat == 0) ? Wq : (mat == 1) ? Wk : Wv;
    Wt[(size_t)mat * UU * DIN + (size_t)n * DIN + k] = (_Float16)W[(size_t)k * UU + n];
}

// ---------------------------------------------------------------------------
// Kernel 1: fused QKV projection.  Y = relu(X @ W + b) in f16.
// grid = (M/64, 3), block = 128 (4 waves).  B-frags straight from the
// pre-transposed Wt in global (cache-resident).  y==2 (V) stores TRANSPOSED
// per batch: Vt[b][u][s].
// ---------------------------------------------------------------------------
__global__ __launch_bounds__(128) void qkv_proj_kernel(
    const float* __restrict__ X, const _Float16* __restrict__ WtAll,
    const float* __restrict__ bq, const float* __restrict__ bk,
    const float* __restrict__ bv,
    _Float16* __restrict__ Qo, _Float16* __restrict__ Ko, _Float16* __restrict__ Vto)
{
    __shared__ _Float16 Xs[64 * 32];    // X tile, row-major [m][k]

    const int tid  = threadIdx.x;
    const int lane = tid & 31, wave = tid >> 5;
    const int lo = lane & 15, hi = lane >> 4;
    const int mbase = blockIdx.x * 64;

    const _Float16* Wt = WtAll + (size_t)blockIdx.y * UU * DIN;   // [n][k]
    const float* bias = (blockIdx.y == 0) ? bq : (blockIdx.y == 1) ? bk : bv;

    const float8 fzero = {};
    float8 acc[8];
#pragma unroll
    for (int nt = 0; nt < 8; ++nt) acc[nt] = fzero;

    for (int kc = 0; kc < DIN / 32; ++kc) {
        __syncthreads();
        // Stage X chunk: 64 rows x 32 cols, f32 -> f16
        const float4* xs = (const float4*)(X + (size_t)mbase * DIN + kc * 32);
#pragma unroll
        for (int i = 0; i < 4; ++i) {
            int e = tid + i * 128;
            int r = e >> 3, c4 = e & 7;
            float4 v = xs[(size_t)r * (DIN / 4) + c4];
            _Float16* d = Xs + r * 32 + c4 * 4;
            d[0] = (_Float16)v.x; d[1] = (_Float16)v.y;
            d[2] = (_Float16)v.z; d[3] = (_Float16)v.w;
        }
        __syncthreads();

        half16 a = load_a_frag(Xs + wave * 16 * 32, 32, lo, hi);
        const _Float16* wrow = Wt + (size_t)lo * DIN + kc * 32 + hi * 16;
#pragma unroll
        for (int nt = 0; nt < 8; ++nt) {
            half16 bf = *(const half16*)(wrow + (size_t)(nt * 16) * DIN);
            acc[nt] = wmma_acc(a, bf, acc[nt]);
        }
    }

    if (blockIdx.y != 2) {
        _Float16* Yout = (blockIdx.y == 0) ? Qo : Ko;   // row-major [m][u]
#pragma unroll
        for (int nt = 0; nt < 8; ++nt) {
            float bb = bias[nt * 16 + lo];
#pragma unroll
            for (int r = 0; r < 8; ++r) {
                float y = fmaxf(acc[nt][r] + bb, 0.0f);
                Yout[(size_t)(mbase + wave * 16 + hi * 8 + r) * UU + nt * 16 + lo] = (_Float16)y;
            }
        }
    } else {
        // Transposed V: Vt[b][u][s]; 8 consecutive s per lane -> one half8 store.
        const int b  = mbase / SS;
        const int s0 = (mbase & (SS - 1)) + wave * 16 + hi * 8;
#pragma unroll
        for (int nt = 0; nt < 8; ++nt) {
            float bb = bias[nt * 16 + lo];
            half8 v8;
#pragma unroll
            for (int r = 0; r < 8; ++r)
                v8[r] = (_Float16)fmaxf(acc[nt][r] + bb, 0.0f);
            *(half8*)(Vto + ((size_t)b * UU + nt * 16 + lo) * SS + s0) = v8;
        }
    }
}

// ---------------------------------------------------------------------------
// Kernel 2: flash attention.  64-key tiles; K async-double-buffered in LDS,
// V B-frags straight from global (transposed layout -> contiguous 32B/lane).
// grid = B * S/64, block = 128 (4 waves, 16 query rows each).
// ---------------------------------------------------------------------------
__global__ __launch_bounds__(128) void flash_attn_kernel(
    const _Float16* __restrict__ Q, const _Float16* __restrict__ K,
    const _Float16* __restrict__ Vt, float* __restrict__ Out)
{
    __shared__ _Float16 Ks[2][64 * 128];   // K tile [t][u], double buffered (2x16KB)
    __shared__ _Float16 Ps[4][16 * 64];    // per-wave P staging (8KB)

    const int tid  = threadIdx.x;
    const int lane = tid & 31, wave = tid >> 5;
    const int lo = lane & 15, hi = lane >> 4;
    const int b  = blockIdx.x >> 6;
    const int qb = blockIdx.x & 63;
    const size_t batch_off = (size_t)b * SS * UU;
    const int qrow = qb * 64 + wave * 16;

    const _Float16* Kg = K + batch_off;                               // [t][u]
    const _Float16* Vg = Vt + batch_off + (size_t)lo * SS + hi * 16;  // per-lane base

    // --- Q fragments: 16 rows x 128, four A-frags, loaded once ---
    half16 qf[4];
    {
        const _Float16* qbase = Q + batch_off + (size_t)(qrow + lo) * UU;
#pragma unroll
        for (int uc = 0; uc < 4; ++uc) {
            half8 a0 = *(const half8*)(qbase + uc * 32 + hi * 16);
            half8 a1 = *(const half8*)(qbase + uc * 32 + hi * 16 + 8);
            half16 f;
#pragma unroll
            for (int i = 0; i < 8; ++i) { f[i] = a0[i]; f[i + 8] = a1[i]; }
            qf[uc] = f;
        }
    }

    const float8 fzero = {};
    float8 o[8];
#pragma unroll
    for (int nt = 0; nt < 8; ++nt) o[nt] = fzero;
    float m[8], l[8];
#pragma unroll
    for (int r = 0; r < 8; ++r) { m[r] = -3.0e38f; l[r] = 0.0f; }

    _Float16* Pw = Ps[wave];

    // Async stage of one 64-key K tile (16KB, 8 x b128 per thread).
    auto stageK = [&](int t0, int bufSel) {
        const _Float16* ks = Kg + (size_t)t0 * UU;
#pragma unroll
        for (int i = 0; i < 8; ++i) {
            int e = (tid + i * 128) * 8;
            async_b128(ks + e, lds_off(&Ks[bufSel][e]));
        }
    };

    stageK(0, 0);
    int buf = 0;

    for (int t0 = 0; t0 < SS; t0 += 64) {
        wait_async0();           // this wave's loads for tile t0 done
        __syncthreads();         // all waves' loads visible; buf^1 free
        if (t0 + 64 < SS) stageK(t0 + 64, buf ^ 1);   // overlap with compute

        const _Float16* Kb = Ks[buf];

        // S = Q K^T : four 16-key sub-tiles, contraction over U=128
        float8 s[4];
#pragma unroll
        for (int c = 0; c < 4; ++c) s[c] = fzero;
#pragma unroll
        for (int uc = 0; uc < 4; ++uc) {
            const _Float16* kr = Kb + uc * 32 + hi * 16 + lo * 128;
#pragma unroll
            for (int c = 0; c < 4; ++c) {
                half16 kb = *(const half16*)(kr + c * 16 * 128);
                s[c] = wmma_acc(qf[uc], kb, s[c]);
            }
        }

        // Online softmax (row M = hi*8+r lives across 16 lanes of a half-wave)
        float p[4][8];
        float8 av;               // per-row rescale factors, packed for V_PK_MUL_F32
#pragma unroll
        for (int r = 0; r < 8; ++r) {
            float v = fmaxf(fmaxf(s[0][r], s[1][r]), fmaxf(s[2][r], s[3][r]));
#pragma unroll
            for (int msk = 1; msk < 16; msk <<= 1) v = fmaxf(v, __shfl_xor(v, msk, 32));
            float mn    = fmaxf(m[r], v);
            float alpha = __expf(m[r] - mn);
            m[r] = mn;
            av[r] = alpha;
            float rs = 0.0f;
#pragma unroll
            for (int c = 0; c < 4; ++c) {
                float e = __expf(s[c][r] - mn);
                p[c][r] = e; rs += e;
            }
#pragma unroll
            for (int msk = 1; msk < 16; msk <<= 1) rs += __shfl_xor(rs, msk, 32);
            l[r] = l[r] * alpha + rs;
        }
        // Rescale running O: vector form -> packed fp32 multiplies
#pragma unroll
        for (int nt = 0; nt < 8; ++nt) o[nt] = o[nt] * av;

        // P: accumulator layout -> A-frag layout via per-wave LDS buffer.
        // DS ops are in-order within a wave: no barrier needed.
#pragma unroll
        for (int r = 0; r < 8; ++r)
#pragma unroll
            for (int c = 0; c < 4; ++c)
                Pw[(hi * 8 + r) * 64 + c * 16 + lo] = (_Float16)p[c][r];

        half16 pf0 = load_a_frag(Pw,      64, lo, hi);   // keys t0+ 0..31
        half16 pf1 = load_a_frag(Pw + 32, 64, lo, hi);   // keys t0+32..63

        // O += P V : V B-frags direct from global transposed layout
        const _Float16* vbase = Vg + t0;
#pragma unroll
        for (int nt = 0; nt < 8; ++nt) {
            half16 v0 = *(const half16*)(vbase + (size_t)(nt * 16) * SS);
            half16 v1 = *(const half16*)(vbase + (size_t)(nt * 16) * SS + 32);
            o[nt] = wmma_acc(pf0, v0, o[nt]);
            o[nt] = wmma_acc(pf1, v1, o[nt]);
        }

        buf ^= 1;
    }

    // Final normalize (packed) + f32 store
    float8 iv;
#pragma unroll
    for (int r = 0; r < 8; ++r) iv[r] = 1.0f / l[r];
#pragma unroll
    for (int nt = 0; nt < 8; ++nt) {
        float8 ov = o[nt] * iv;
#pragma unroll
        for (int r = 0; r < 8; ++r) {
            Out[batch_off + (size_t)(qrow + hi * 8 + r) * UU + nt * 16 + lo] = ov[r];
        }
    }
}

extern "C" void kernel_launch(void* const* d_in, const int* in_sizes, int n_in,
                              void* d_out, int out_size, void* d_ws, size_t ws_size,
                              hipStream_t stream) {
    const float* X  = (const float*)d_in[0];
    const float* Wq = (const float*)d_in[1];
    const float* bq = (const float*)d_in[2];
    const float* Wk = (const float*)d_in[3];
    const float* bk = (const float*)d_in[4];
    const float* Wv = (const float*)d_in[5];
    const float* bv = (const float*)d_in[6];

    const size_t proj_elems = (size_t)BB * SS * UU;     // 2M halves each
    _Float16* Qw  = (_Float16*)d_ws;
    _Float16* Kw  = Qw + proj_elems;
    _Float16* Vtw = Kw + proj_elems;                    // transposed: [b][u][s]
    _Float16* Wtw = Vtw + proj_elems;                   // [3][128][512] f16

    // 0) Pre-transpose weights (once, tiny).
    w_transpose_kernel<<<dim3(3 * 65536 / 256), 256, 0, stream>>>(Wq, Wk, Wv, Wtw);

    // 1) Q/K/V = relu(X @ W + b), f16 (V transposed).
    qkv_proj_kernel<<<dim3((BB * SS) / 64, 3), 128, 0, stream>>>(
        X, Wtw, bq, bk, bv, Qw, Kw, Vtw);

    // 2) Flash attention: O = softmax(Q K^T) V.
    flash_attn_kernel<<<dim3((BB * SS) / 64), 128, 0, stream>>>(
        Qw, Kw, Vtw, (float*)d_out);
}
